// energy_latencyW_50_54176717471926
// MI455X (gfx1250) — matrix operand order repaired and checked by reading, not compiled, not run on hardware
//
#include <hip/hip_runtime.h>
#include <hip/hip_bf16.h>
#include <math.h>

// ---------------------------------------------------------------------------
// Types for CDNA5 WMMA (wave32): V_WMMA_F32_16X16X32_BF16
// ---------------------------------------------------------------------------
typedef __bf16 v16bf __attribute__((ext_vector_type(16)));
typedef __bf16 bf16x8 __attribute__((ext_vector_type(8)));
typedef float  v8f    __attribute__((ext_vector_type(8)));

__device__ __forceinline__ v16bf cat8(bf16x8 lo, bf16x8 hi) {
  return __builtin_shufflevector(lo, hi, 0,1,2,3,4,5,6,7,8,9,10,11,12,13,14,15);
}

__device__ __forceinline__ float softplus_f(float v) {
  return fmaxf(v, 0.f) + log1pf(expf(-fabsf(v)));   // stable softplus
}

// CDNA5 async copy: global -> LDS, 16B per lane, tracked by ASYNCcnt.
__device__ __forceinline__ void async_ld_b128(unsigned lds_off, const void* g) {
  asm volatile("global_load_async_to_lds_b128 %0, %1, off"
               :: "v"(lds_off), "v"(g) : "memory");
}
__device__ __forceinline__ void wait_async0() {
  asm volatile("s_wait_asynccnt 0x0" ::: "memory");
}

// ---------------------------------------------------------------------------
// Problem constants
// ---------------------------------------------------------------------------
#define NB     16384
#define NNODES 200
#define NEDGE  800
#define XCOLS  2600      // 50*52
#define FEAT   2800
#define KPAD   2816      // 88 chunks of 32
#define NKC1   88
#define KCFAST 81        // chunks [0,81) read only x (80*32+31 = 2591 < 2600)
#define MT     64        // rows per block
#define NGRID  (NB / MT) // 256 blocks

// workspace layout (bytes)
#define OFF_W1  0
#define OFF_W2  720896           // 2816*128*2
#define OFF_W3  753664           // + 128*128*2
#define OFF_NS  770048           // + 128*64*2
#define OFF_ND  770848
#define OFF_HWT 771648

// ---------------------------------------------------------------------------
// Prep: degree norms, hw table, weight conversion to bf16 (W1 zero-padded)
// ---------------------------------------------------------------------------
__global__ __launch_bounds__(256) void prep_kernel(
    const int* __restrict__ esrc, const int* __restrict__ edst,
    const float* __restrict__ emb, const float* __restrict__ gcw,
    const float* __restrict__ W1, const float* __restrict__ W2,
    const float* __restrict__ W3,
    __bf16* __restrict__ W1bf, __bf16* __restrict__ W2bf,
    __bf16* __restrict__ W3bf,
    float* __restrict__ nsrcO, float* __restrict__ ndstO,
    float* __restrict__ hwtO)
{
  const int g = blockIdx.x * blockDim.x + threadIdx.x;
  const int total = gridDim.x * blockDim.x;
  for (int i = g; i < KPAD * 128; i += total) {
    int k = i >> 7;
    W1bf[i] = (k < FEAT) ? (__bf16)W1[i] : (__bf16)0.f;
  }
  for (int i = g; i < 128 * 128; i += total) W2bf[i] = (__bf16)W2[i];
  for (int i = g; i < 128 * 64;  i += total) W3bf[i] = (__bf16)W3[i];

  if (blockIdx.x == 0) {
    __shared__ int dO[NNODES], dI[NNODES];
    const int t = threadIdx.x;
    if (t < NNODES) { dO[t] = 0; dI[t] = 0; }
    __syncthreads();
    for (int e = t; e < NEDGE; e += blockDim.x) {
      atomicAdd(&dO[esrc[e]], 1);
      atomicAdd(&dI[edst[e]], 1);
    }
    __syncthreads();
    if (t < NNODES) {
      int o = dO[t] < 1 ? 1 : dO[t];
      int i = dI[t] < 1 ? 1 : dI[t];
      nsrcO[t] = 1.f / sqrtf((float)o);
      ndstO[t] = 1.f / sqrtf((float)i);
    }
    if (t < 3) {
      float s = 0.f;
      for (int j = 0; j < 5; j++) s += emb[t * 5 + j] * gcw[j];
      hwtO[t] = s;
    }
  }
}

// ---------------------------------------------------------------------------
// Fused: graph-conv + (feat@W1 softplus) + (@W2 softplus) + (@W3 tanhshrink)
//        + (@W4 sigmoid).  64 rows per block, 8 waves (wave32).
// ---------------------------------------------------------------------------
__global__ __launch_bounds__(256, 1) void mlp_kernel(
    const float* __restrict__ x, const int* __restrict__ apps,
    const int* __restrict__ esrc, const int* __restrict__ edst,
    const float* __restrict__ gcb,
    const float* __restrict__ b1, const float* __restrict__ b2,
    const float* __restrict__ b3,
    const float* __restrict__ W4, const float* __restrict__ b4,
    const __bf16* __restrict__ W1bf, const __bf16* __restrict__ W2bf,
    const __bf16* __restrict__ W3bf,
    const float* __restrict__ nsrcG, const float* __restrict__ ndstG,
    const float* __restrict__ hwtG,
    float* __restrict__ out)
{
  // --- LDS ---------------------------------------------------------------
  __shared__ float gBuf[MT * NNODES];                               // 51200 B
  __shared__ __attribute__((aligned(16))) __bf16 Abuf[2 * MT * 40]; // 10240 B (double buf)
  __shared__ __attribute__((aligned(16))) __bf16 Bbuf[2 * 32 * 144];// 18432 B (double buf)
  __shared__ uint4 dynbuf4[53504 / 16];                             // 53504 B (phase union)
  __shared__ float nsrcS[NNODES], ndstS[NNODES], hwtS[3];

  char* dynbuf = (char*)dynbuf4;
  // phase A view
  unsigned char* appsB = (unsigned char*)dynbuf;        // [64][200] u8
  short* esrcS  = (short*)(dynbuf + 12800);             // [800]
  short* edstS  = (short*)(dynbuf + 14400);             // [800]
  float* ecoefS = (float*)(dynbuf + 16000);             // [800]
  // phase B view
  __bf16* a1 = (__bf16*)dynbuf;                         // [64][144] bf16
  __bf16* a2 = (__bf16*)(dynbuf + 18432);               // [64][144] bf16
  float*  a3 = (float*)(dynbuf + 36864);                // [64][65]  f32

  const int tid = threadIdx.x;
  const int rowBase = blockIdx.x * MT;

  // ======================= Phase A: graph conv ===========================
  if (tid < NNODES) { nsrcS[tid] = nsrcG[tid]; ndstS[tid] = ndstG[tid]; }
  if (tid < 3) hwtS[tid] = hwtG[tid];
  for (int i = tid; i < MT * NNODES; i += 256) {
    gBuf[i] = 0.f;
    appsB[i] = (unsigned char)apps[(size_t)rowBase * NNODES + i];
  }
  __syncthreads();
  for (int e = tid; e < NEDGE; e += 256) {
    int s = esrc[e];
    esrcS[e] = (short)s;
    edstS[e] = (short)edst[e];
    ecoefS[e] = nsrcS[s];
  }
  __syncthreads();
  if (tid < MT) {                       // one thread per row: deterministic
    float* acc = &gBuf[tid * NNODES];
    const unsigned char* ar = &appsB[tid * NNODES];
    for (int e = 0; e < NEDGE; e++)
      acc[edstS[e]] += ecoefS[e] * hwtS[ar[esrcS[e]]];
  }
  __syncthreads();
  {
    const float gb = gcb[0];
    for (int i = tid; i < MT * NNODES; i += 256) {
      int n = i % NNODES;
      gBuf[i] = gBuf[i] * ndstS[n] + gb;
    }
  }
  __syncthreads();

  // ======================= WMMA fragment coords ==========================
  const int wid = tid >> 5, lane = tid & 31;
  const int waveM = wid & 3;        // 4 row-waves x 16 rows
  const int waveN = wid >> 2;       // 2 col-waves
  const int mrow  = waveM * 16 + (lane & 15);
  const int kbase = (lane >> 4) * 8;            // lanes 16..31 hold k 8..15 / 24..31
  const int lrow  = (lane >> 4);
  // A-stage loader coords
  const int ar_r = tid >> 2, ar_q = tid & 3;
  const float* xrow = x + (size_t)(rowBase + ar_r) * XCOLS;
  const float* grow = &gBuf[ar_r * NNODES];
  // B-stage (async) coords: 32 rows x 8 segs of 16 bf16 (2 x 16B)
  const int br = tid >> 3, bs = tid & 7;
  unsigned ldsB[2];
  ldsB[0] = (unsigned)(size_t)&Bbuf[br * 144 + bs * 16];
  ldsB[1] = (unsigned)(size_t)&Bbuf[32 * 144 + br * 144 + bs * 16];

  // ======================= GEMM1: feat(2816) @ W1 -> 128 =================
  // Single-barrier software pipeline: A prefetched via regs (+cvt),
  // B streamed via global_load_async_to_lds_b128 into alternating buffers.
  v8f c1[4] = {};
  bf16x8 apre;
  // ---- prologue: chunk 0 (always fast path) ----
  {
    const int k0 = ar_q * 8;
    float4 f0 = *(const float4*)(xrow + k0);
    float4 f1 = *(const float4*)(xrow + k0 + 4);
    apre[0]=(__bf16)f0.x; apre[1]=(__bf16)f0.y; apre[2]=(__bf16)f0.z; apre[3]=(__bf16)f0.w;
    apre[4]=(__bf16)f1.x; apre[5]=(__bf16)f1.y; apre[6]=(__bf16)f1.z; apre[7]=(__bf16)f1.w;
  }
  async_ld_b128(ldsB[0], &W1bf[(size_t)br * 128 + bs * 16]);
  async_ld_b128(ldsB[0] + 16, &W1bf[(size_t)br * 128 + bs * 16 + 8]);

  for (int kc = 0; kc < NKC1; kc++) {
    const int p = kc & 1;
    // 1) commit prefetched A chunk kc to LDS
    *(bf16x8*)&Abuf[p * (MT * 40) + ar_r * 40 + ar_q * 8] = apre;
    // 2) drain async B chunk kc, then barrier
    wait_async0();
    __syncthreads();
    // 3) issue async B chunk kc+1 (other buffer: readers finished pre-barrier)
    const int kcn = kc + 1;
    if (kcn < NKC1) {
      async_ld_b128(ldsB[kcn & 1],
                    &W1bf[(size_t)(kcn * 32 + br) * 128 + bs * 16]);
      async_ld_b128(ldsB[kcn & 1] + 16,
                    &W1bf[(size_t)(kcn * 32 + br) * 128 + bs * 16 + 8]);
      // 4) prefetch A chunk kc+1 into regs (overlaps with WMMAs below)
      if (kcn < KCFAST) {               // uniform (scalar) branch: pure-x chunks
        const int k0 = kcn * 32 + ar_q * 8;
        float4 f0 = *(const float4*)(xrow + k0);
        float4 f1 = *(const float4*)(xrow + k0 + 4);
        apre[0]=(__bf16)f0.x; apre[1]=(__bf16)f0.y; apre[2]=(__bf16)f0.z; apre[3]=(__bf16)f0.w;
        apre[4]=(__bf16)f1.x; apre[5]=(__bf16)f1.y; apre[6]=(__bf16)f1.z; apre[7]=(__bf16)f1.w;
      } else {                          // seam/g/pad chunks (7 of 88)
        const int k0 = kcn * 32 + ar_q * 8;
        #pragma unroll
        for (int i = 0; i < 8; i++) {
          int k = k0 + i;
          float v = 0.f;
          if (k < XCOLS)      v = xrow[k];
          else if (k < FEAT)  v = grow[k - XCOLS];
          apre[i] = (__bf16)v;
        }
      }
    }
    // 5) WMMAs on chunk kc
    const __bf16* Ab = &Abuf[p * (MT * 40)];
    const __bf16* Bb = &Bbuf[p * (32 * 144)];
    bf16x8 alo = *(const bf16x8*)&Ab[mrow * 40 + kbase];
    bf16x8 ahi = *(const bf16x8*)&Ab[mrow * 40 + kbase + 16];
    v16bf af = cat8(alo, ahi);
    #pragma unroll
    for (int j = 0; j < 4; j++) {
      int n0 = waveN * 64 + j * 16;
      v16bf bf;
      #pragma unroll
      for (int s = 0; s < 16; s++)
        bf[s] = Bb[(kbase + (s & 7) + ((s >> 3) << 4)) * 144 + n0 + (lane & 15)];
      c1[j] = __builtin_amdgcn_wmma_f32_16x16x32_bf16(
          false, af, false, bf, (short)0, c1[j], false, false);
    }
  }
  __syncthreads();     // protect dynbuf reuse (a1) and Bbuf reuse below
  // bias + softplus -> a1 (bf16)
  #pragma unroll
  for (int j = 0; j < 4; j++) {
    int n = waveN * 64 + j * 16 + (lane & 15);
    float bias = b1[n];
    int mb = waveM * 16 + (lrow << 3);
    #pragma unroll
    for (int i = 0; i < 8; i++)
      a1[(mb + i) * 144 + n] = (__bf16)softplus_f(c1[j][i] + bias);
  }
  __syncthreads();

  // ======================= GEMM2: 128 @ W2 -> 128 ========================
  v8f c2[4] = {};
  for (int kc = 0; kc < 4; kc++) {
    *(uint4*)&Bbuf[br * 144 + bs * 16] =
        *(const uint4*)&W2bf[(kc * 32 + br) * 128 + bs * 16];
    *(uint4*)&Bbuf[br * 144 + bs * 16 + 8] =
        *(const uint4*)&W2bf[(kc * 32 + br) * 128 + bs * 16 + 8];
    __syncthreads();
    bf16x8 alo = *(const bf16x8*)&a1[mrow * 144 + kc * 32 + kbase];
    bf16x8 ahi = *(const bf16x8*)&a1[mrow * 144 + kc * 32 + kbase + 16];
    v16bf af = cat8(alo, ahi);
    #pragma unroll
    for (int j = 0; j < 4; j++) {
      int n0 = waveN * 64 + j * 16;
      v16bf bf;
      #pragma unroll
      for (int s = 0; s < 16; s++)
        bf[s] = Bbuf[(kbase + (s & 7) + ((s >> 3) << 4)) * 144 + n0 + (lane & 15)];
      c2[j] = __builtin_amdgcn_wmma_f32_16x16x32_bf16(
          false, af, false, bf, (short)0, c2[j], false, false);
    }
    __syncthreads();
  }
  #pragma unroll
  for (int j = 0; j < 4; j++) {
    int n = waveN * 64 + j * 16 + (lane & 15);
    float bias = b2[n];
    int mb = waveM * 16 + (lrow << 3);
    #pragma unroll
    for (int i = 0; i < 8; i++)
      a2[(mb + i) * 144 + n] = (__bf16)softplus_f(c2[j][i] + bias);
  }
  __syncthreads();

  // ======================= GEMM3: 128 @ W3 -> 64, tanhshrink =============
  v8f c3[2] = {};
  for (int kc = 0; kc < 4; kc++) {
    *(uint4*)&Bbuf[br * 144 + bs * 8] =
        *(const uint4*)&W3bf[(kc * 32 + br) * 64 + bs * 8];
    __syncthreads();
    bf16x8 alo = *(const bf16x8*)&a2[mrow * 144 + kc * 32 + kbase];
    bf16x8 ahi = *(const bf16x8*)&a2[mrow * 144 + kc * 32 + kbase + 16];
    v16bf af = cat8(alo, ahi);
    #pragma unroll
    for (int j = 0; j < 2; j++) {
      int n0 = waveN * 32 + j * 16;
      v16bf bf;
      #pragma unroll
      for (int s = 0; s < 16; s++)
        bf[s] = Bbuf[(kbase + (s & 7) + ((s >> 3) << 4)) * 144 + n0 + (lane & 15)];
      c3[j] = __builtin_amdgcn_wmma_f32_16x16x32_bf16(
          false, af, false, bf, (short)0, c3[j], false, false);
    }
    __syncthreads();
  }
  #pragma unroll
  for (int j = 0; j < 2; j++) {
    int n = waveN * 32 + j * 16 + (lane & 15);
    float bias = b3[n];
    int mb = waveM * 16 + (lrow << 3);
    #pragma unroll
    for (int i = 0; i < 8; i++) {
      float v = c3[j][i] + bias;
      a3[(mb + i) * 65 + n] = v - tanhf(v);     // Tanhshrink
    }
  }
  __syncthreads();

  // ======================= GEMM4: 64 @ W4 -> 2, sigmoid ==================
  if (tid < 2 * MT) {
    int rr = tid >> 1, cc = tid & 1;
    float s = b4[cc];
    const float* ar = &a3[rr * 65];
    #pragma unroll 8
    for (int k = 0; k < 64; k++) s += ar[k] * W4[k * 2 + cc];
    out[(size_t)(rowBase + rr) * 2 + cc] = 1.f / (1.f + expf(-s));
  }
}

// ---------------------------------------------------------------------------
// Host entry
// ---------------------------------------------------------------------------
extern "C" void kernel_launch(void* const* d_in, const int* in_sizes, int n_in,
                              void* d_out, int out_size, void* d_ws, size_t ws_size,
                              hipStream_t stream) {
  (void)in_sizes; (void)n_in; (void)out_size; (void)ws_size;
  const float* x    = (const float*)d_in[0];
  const int*   apps = (const int*)d_in[1];
  const int*   esrc = (const int*)d_in[2];
  const int*   edst = (const int*)d_in[3];
  const float* emb  = (const float*)d_in[4];
  const float* gcw  = (const float*)d_in[5];
  const float* gcb  = (const float*)d_in[6];
  const float* W1   = (const float*)d_in[7];
  const float* b1   = (const float*)d_in[8];
  const float* W2   = (const float*)d_in[9];
  const float* b2   = (const float*)d_in[10];
  const float* W3   = (const float*)d_in[11];
  const float* b3   = (const float*)d_in[12];
  const float* W4   = (const float*)d_in[13];
  const float* b4   = (const float*)d_in[14];
  float* out = (float*)d_out;

  char* ws = (char*)d_ws;
  __bf16* W1bf = (__bf16*)(ws + OFF_W1);
  __bf16* W2bf = (__bf16*)(ws + OFF_W2);
  __bf16* W3bf = (__bf16*)(ws + OFF_W3);
  float*  nsrc = (float*)(ws + OFF_NS);
  float*  ndst = (float*)(ws + OFF_ND);
  float*  hwt  = (float*)(ws + OFF_HWT);

  prep_kernel<<<256, 256, 0, stream>>>(esrc, edst, emb, gcw, W1, W2, W3,
                                       W1bf, W2bf, W3bf, nsrc, ndst, hwt);
  mlp_kernel<<<NGRID, 256, 0, stream>>>(x, apps, esrc, edst, gcb,
                                        b1, b2, b3, W4, b4,
                                        W1bf, W2bf, W3bf,
                                        nsrc, ndst, hwt, out);
}
